// Attention_73864847556918
// MI455X (gfx1250) — matrix operand order repaired and checked by reading, not compiled, not run on hardware
//
#include <hip/hip_runtime.h>
#include <hip/hip_bf16.h>

// ---------------------------------------------------------------------------
// MI455X (gfx1250) fused vision-attention block.
// bf16 WMMA (v_wmma_f32_16x16x32_bf16) + f32 accum for all GEMMs; f32 RoPE /
// softmax. Flash attention with cooperative LDS staging of K/V tiles via
// gfx1250 async-to-LDS copies (double buffered, s_wait_asynccnt pipelined).
// ---------------------------------------------------------------------------

typedef __bf16 bf16_t;
typedef __attribute__((ext_vector_type(16))) __bf16 v16bf;
typedef __attribute__((ext_vector_type(8)))  __bf16 v8bf;
typedef __attribute__((ext_vector_type(4)))  __bf16 v4bf;
typedef __attribute__((ext_vector_type(8)))  float  v8f;
typedef __attribute__((ext_vector_type(4)))  int    v4i;

// --- WMMA wrapper ----------------------------------------------------------
__device__ __forceinline__ v8f wmma_bf16(v16bf a, v16bf b, v8f c) {
    return __builtin_amdgcn_wmma_f32_16x16x32_bf16(
        /*neg_a=*/false, a, /*neg_b=*/false, b,
        /*c_mod=*/(short)0, c, /*reuse_a=*/false, /*reuse_b=*/false);
}

// A-fragment (16x32, MxK): lane (r = lane&15, hi = lane>>4) holds
// K = k0 + hi*8 + [0,8) and k0 + 16 + hi*8 + [0,8) of row r.
__device__ __forceinline__ v16bf load_a_frag(const bf16_t* rowp, int k0, int hi) {
    union { v16bf v; v8bf h[2]; } u;
    u.h[0] = *(const v8bf*)(rowp + k0 + hi * 8);
    u.h[1] = *(const v8bf*)(rowp + k0 + 16 + hi * 8);
    return u.v;
}

// B-fragment (32x16, KxN) loaded column-major: lane (n = lane&15, hi) holds
// K = k0 + hi*8 + [0,8) and k0 + 16 + hi*8 + [0,8) of column n (== row n of the row-major operand).
__device__ __forceinline__ v16bf load_b_frag(const bf16_t* rowp, int k0, int hi) {
    union { v16bf v; v8bf h[2]; } u;
    u.h[0] = *(const v8bf*)(rowp + k0 + hi * 8);
    u.h[1] = *(const v8bf*)(rowp + k0 + 16 + hi * 8);
    return u.v;
}

// --- gfx1250 async global->LDS copy (16B per lane), with safe fallback -----
__device__ __forceinline__ void async_cp16(bf16_t* lds, const bf16_t* gmem) {
#if __has_builtin(__builtin_amdgcn_global_load_async_to_lds_b128)
    __builtin_amdgcn_global_load_async_to_lds_b128(
        (__attribute__((address_space(1))) v4i*)gmem,
        (__attribute__((address_space(3))) v4i*)lds, 0, 0);
#else
    *(v8bf*)lds = *(const v8bf*)gmem;      // synchronous fallback
#endif
}

__device__ __forceinline__ void wait_async0() {
#if __has_builtin(__builtin_amdgcn_s_wait_asynccnt)
    __builtin_amdgcn_s_wait_asynccnt(0);
#elif __has_builtin(__builtin_amdgcn_global_load_async_to_lds_b128)
    asm volatile("s_wait_asynccnt 0x0" ::: "memory");
#endif
}

// --- f32 -> bf16 bulk convert (4 elems/thread) -----------------------------
__global__ __launch_bounds__(256)
void k_f32_to_bf16(const float4* __restrict__ in, bf16_t* __restrict__ out, int n4) {
    int i = blockIdx.x * 256 + threadIdx.x;
    if (i < n4) {
        float4 f = in[i];
        v4bf o;
        o[0] = (bf16_t)f.x; o[1] = (bf16_t)f.y;
        o[2] = (bf16_t)f.z; o[3] = (bf16_t)f.w;
        *(v4bf*)(out + 4 * (size_t)i) = o;
    }
}

// --- RoPE tables: cos/sin [256][64] ----------------------------------------
__global__ __launch_bounds__(256)
void k_rope_tables(float* __restrict__ cosT, float* __restrict__ sinT) {
    int i = blockIdx.x * 256 + threadIdx.x;      // 0..16383
    int s = i >> 6, d = i & 63;
    int hh = s >> 4, ww = s & 15;
    int dd = (d < 32) ? d : (d - 32);
    float pos = (d < 32) ? (float)hh : (float)ww;
    int j = dd >> 1;
    float freq = powf(10000.0f, -((float)j) / 16.0f);
    float ang = pos * freq;
    cosT[i] = cosf(ang);
    sinT[i] = sinf(ang);
}

// ===========================================================================
// Stage 1/3 shared GEMM body: C[M,NC] = A[M,1024] * W[NC,1024]^T
// 128x128 block tile, 8 waves (2x4), LDS-staged A/B tiles, double buffered
// async copies. Epilogue differs -> two kernels below.
// ===========================================================================
#define GEMM_TILE_STRIDE 40   // 32 + 8 pad (bf16 elems), keeps 16B alignment

__device__ __forceinline__ void gemm_stage_tiles(
        bf16_t* At, bf16_t* Bt,
        const bf16_t* __restrict__ A, const bf16_t* __restrict__ W,
        int mBase0, int nBase0, int k0, int tid) {
    const int C = 1024;
    int row = tid >> 1;            // 0..127
    int half = (tid & 1) * 16;     // 0 or 16 (bf16 elems)
    const bf16_t* ga = A + (size_t)(mBase0 + row) * C + k0 + half;
    const bf16_t* gb = W + (size_t)(nBase0 + row) * C + k0 + half;
    bf16_t* la = At + row * GEMM_TILE_STRIDE + half;
    bf16_t* lb = Bt + row * GEMM_TILE_STRIDE + half;
    async_cp16(la,     ga);
    async_cp16(la + 8, ga + 8);
    async_cp16(lb,     gb);
    async_cp16(lb + 8, gb + 8);
}

// Computes acc[4][2] for this wave; shared by both GEMM kernels.
__device__ __forceinline__ void gemm_core(
        v8f acc[4][2],
        const bf16_t* __restrict__ A, const bf16_t* __restrict__ W,
        int mBase0, int nBase0, int waveM, int waveN, int lr, int hi, int tid,
        bf16_t* AtBuf, bf16_t* BtBuf) {   // each [2][128*GEMM_TILE_STRIDE]
    const int C = 1024;
#pragma unroll
    for (int i = 0; i < 4; i++)
#pragma unroll
        for (int j = 0; j < 2; j++) acc[i][j] = v8f{};

    gemm_stage_tiles(AtBuf, BtBuf, A, W, mBase0, nBase0, 0, tid);

    for (int k0 = 0; k0 < C; k0 += 32) {
        int buf = (k0 >> 5) & 1;
        bf16_t* At = AtBuf + buf * (128 * GEMM_TILE_STRIDE);
        bf16_t* Bt = BtBuf + buf * (128 * GEMM_TILE_STRIDE);
        wait_async0();
        __syncthreads();
        if (k0 + 32 < C) {
            int nb = buf ^ 1;
            gemm_stage_tiles(AtBuf + nb * (128 * GEMM_TILE_STRIDE),
                             BtBuf + nb * (128 * GEMM_TILE_STRIDE),
                             A, W, mBase0, nBase0, k0 + 32, tid);
        }
        v16bf af[4], bfr[2];
#pragma unroll
        for (int i = 0; i < 4; i++)
            af[i] = load_a_frag(At + (waveM * 64 + i * 16 + lr) * GEMM_TILE_STRIDE, 0, hi);
#pragma unroll
        for (int j = 0; j < 2; j++)
            bfr[j] = load_b_frag(Bt + (waveN * 32 + j * 16 + lr) * GEMM_TILE_STRIDE, 0, hi);
#pragma unroll
        for (int i = 0; i < 4; i++)
#pragma unroll
            for (int j = 0; j < 2; j++)
                acc[i][j] = wmma_bf16(af[i], bfr[j], acc[i][j]);
    }
}

// --- Stage 1: QKV GEMM + bias + RoPE, scatter to Q/K [B,H,N,D], Vt [B,H,D,N]
__global__ __launch_bounds__(256)
void k_qkv_gemm(const bf16_t* __restrict__ X,     // [4096,1024] bf16
                const bf16_t* __restrict__ W,     // [3072,1024] bf16
                const float*  __restrict__ bias,  // [3072]
                const float*  __restrict__ cosT,
                const float*  __restrict__ sinT,
                bf16_t* __restrict__ Qb, bf16_t* __restrict__ Kb,
                bf16_t* __restrict__ Vt) {
    __shared__ bf16_t AtBuf[2][128 * GEMM_TILE_STRIDE];
    __shared__ bf16_t BtBuf[2][128 * GEMM_TILE_STRIDE];

    int tid = threadIdx.x;
    int nBlk = blockIdx.x, mBlk = blockIdx.y;
    int wave = tid >> 5, lane = tid & 31;
    int lr = lane & 15, hi = lane >> 4;
    int waveM = wave >> 2, waveN = wave & 3;
    int mBase = mBlk * 128 + waveM * 64;
    int nBase = nBlk * 128 + waveN * 32;

    v8f acc[4][2];
    gemm_core(acc, X, W, mBlk * 128, nBlk * 128, waveM, waveN, lr, hi, tid,
              &AtBuf[0][0], &BtBuf[0][0]);

    // Epilogue: bias, RoPE (pair exchange across adjacent lanes), scatter.
#pragma unroll
    for (int i = 0; i < 4; i++) {
#pragma unroll
        for (int j = 0; j < 2; j++) {
#pragma unroll
            for (int jj = 0; jj < 8; jj++) {
                int row = mBase + i * 16 + hi * 8 + jj;    // global token row
                int col = nBase + j * 16 + lr;             // 0..3071
                float v = acc[i][j][jj] + bias[col];
                int which = col >> 10;                     // 0=q,1=k,2=v
                int hd = col & 1023;
                int h = hd >> 6, d = hd & 63;
                int b = row >> 11, n = row & 2047, s = n & 255;
                float partner = __shfl_xor(v, 1, 32);      // d and d^1 are lane,lane^1
                if (which < 2) {
                    float rot = (d & 1) ? partner : -partner;
                    v = v * cosT[s * 64 + d] + rot * sinT[s * 64 + d];
                }
                size_t bh = (size_t)(b * 16 + h);
                if (which == 0)       Qb[(bh * 2048 + n) * 64 + d] = (bf16_t)v;
                else if (which == 1)  Kb[(bh * 2048 + n) * 64 + d] = (bf16_t)v;
                else                  Vt[(bh * 64 + d) * 2048 + n] = (bf16_t)v;
            }
        }
    }
}

// --- Stage 3: output projection, f32 result --------------------------------
__global__ __launch_bounds__(256)
void k_proj_gemm(const bf16_t* __restrict__ A,    // [4096,1024] bf16
                 const bf16_t* __restrict__ W,    // [1024,1024] bf16
                 const float*  __restrict__ bias, // [1024]
                 float* __restrict__ out) {       // [4096,1024] f32
    __shared__ bf16_t AtBuf[2][128 * GEMM_TILE_STRIDE];
    __shared__ bf16_t BtBuf[2][128 * GEMM_TILE_STRIDE];

    const int C = 1024;
    int tid = threadIdx.x;
    int nBlk = blockIdx.x, mBlk = blockIdx.y;
    int wave = tid >> 5, lane = tid & 31;
    int lr = lane & 15, hi = lane >> 4;
    int waveM = wave >> 2, waveN = wave & 3;
    int mBase = mBlk * 128 + waveM * 64;
    int nBase = nBlk * 128 + waveN * 32;

    v8f acc[4][2];
    gemm_core(acc, A, W, mBlk * 128, nBlk * 128, waveM, waveN, lr, hi, tid,
              &AtBuf[0][0], &BtBuf[0][0]);

#pragma unroll
    for (int i = 0; i < 4; i++) {
#pragma unroll
        for (int j = 0; j < 2; j++) {
#pragma unroll
            for (int jj = 0; jj < 8; jj++) {
                int row = mBase + i * 16 + hi * 8 + jj;
                int col = nBase + j * 16 + lr;
                out[(size_t)row * C + col] = acc[i][j][jj] + bias[col];
            }
        }
    }
}

// ===========================================================================
// Stage 2: flash attention. One block per (b, h, 128 q-rows); 8 waves.
// K (32x64) and V^T (64x32) tiles staged in LDS once per block per k-tile by
// async copies (double buffered); all 8 waves consume them (8x traffic cut).
// ===========================================================================
#define KT_STRIDE 72   // 64 + 8 pad
#define VT_STRIDE 40   // 32 + 8 pad

__device__ __forceinline__ void flash_stage_kv(
        bf16_t* Kt, bf16_t* Vtile,
        const bf16_t* __restrict__ Kp, const bf16_t* __restrict__ Vp,
        int kt, int tid) {
    {   // K tile: rows kt..kt+31, cols d 0..63  (one 16B chunk per thread)
        int row = tid >> 3, seg = (tid & 7) * 8;
        async_cp16(Kt + row * KT_STRIDE + seg,
                   Kp + (size_t)(kt + row) * 64 + seg);
    }
    {   // V^T tile: rows d 0..63, cols kt..kt+31 (one 16B chunk per thread)
        int row = tid >> 2, seg = (tid & 3) * 8;
        async_cp16(Vtile + row * VT_STRIDE + seg,
                   Vp + (size_t)row * 2048 + kt + seg);
    }
}

__global__ __launch_bounds__(256)
void k_flash_attn(const bf16_t* __restrict__ Qb, const bf16_t* __restrict__ Kb,
                  const bf16_t* __restrict__ Vt, const float* __restrict__ mask,
                  bf16_t* __restrict__ O) {
    const int Ntok = 2048;
    int tid = threadIdx.x;
    int qBlk = blockIdx.x, h = blockIdx.y, b = blockIdx.z;
    int wave = tid >> 5, lane = tid & 31;
    int lr = lane & 15, hi = lane >> 4;
    size_t bh = (size_t)(b * 16 + h);
    const bf16_t* Qp = Qb + bh * 2048 * 64;
    const bf16_t* Kp = Kb + bh * 2048 * 64;
    const bf16_t* Vp = Vt + bh * 64 * 2048;
    int q0 = qBlk * 128 + wave * 16;

    __shared__ bf16_t KtBuf[2][32 * KT_STRIDE];
    __shared__ bf16_t VtBuf[2][64 * VT_STRIDE];
    __shared__ bf16_t pbuf[8][16 * 32];     // per-wave P tile (row-major 16x32)
    bf16_t* pw = pbuf[wave];

    v16bf qf0 = load_a_frag(Qp + (size_t)(q0 + lr) * 64, 0, hi);
    v16bf qf1 = load_a_frag(Qp + (size_t)(q0 + lr) * 64, 32, hi);

    v8f z = {};
    v8f oacc[4];
#pragma unroll
    for (int t = 0; t < 4; t++) oacc[t] = z;
    float mI[8], lI[8];
#pragma unroll
    for (int jj = 0; jj < 8; jj++) { mI[jj] = -3.0e38f; lI[jj] = 0.0f; }

    const float scale = 0.125f;             // 64^-0.5
    const float* maskB = mask + (size_t)b * 2048 * 2048;

    flash_stage_kv(KtBuf[0], VtBuf[0], Kp, Vp, 0, tid);

    for (int kt = 0; kt < Ntok; kt += 32) {
        int buf = (kt >> 5) & 1;
        bf16_t* Kt = KtBuf[buf];
        bf16_t* Vtile = VtBuf[buf];
        wait_async0();
        __syncthreads();
        if (kt + 32 < Ntok)
            flash_stage_kv(KtBuf[buf ^ 1], VtBuf[buf ^ 1], Kp, Vp, kt + 32, tid);

        v16bf kf00 = load_b_frag(Kt + lr * KT_STRIDE, 0, hi);
        v16bf kf01 = load_b_frag(Kt + lr * KT_STRIDE, 32, hi);
        v16bf kf10 = load_b_frag(Kt + (16 + lr) * KT_STRIDE, 0, hi);
        v16bf kf11 = load_b_frag(Kt + (16 + lr) * KT_STRIDE, 32, hi);

        v8f s0 = wmma_bf16(qf0, kf00, z);
        s0 = wmma_bf16(qf1, kf01, s0);
        v8f s1 = wmma_bf16(qf0, kf10, z);
        s1 = wmma_bf16(qf1, kf11, s1);

        float p0[8], p1[8], alpha[8];
#pragma unroll
        for (int jj = 0; jj < 8; jj++) {
            int gq = q0 + hi * 8 + jj;
            float m0 = maskB[(size_t)gq * 2048 + kt + lr];
            float m1 = maskB[(size_t)gq * 2048 + kt + 16 + lr];
            float b0 = (m0 == 0.f) ? -1.0e8f : ((m0 == 1.f) ? 0.f : m0);
            float b1 = (m1 == 0.f) ? -1.0e8f : ((m1 == 1.f) ? 0.f : m1);
            float v0 = s0[jj] * scale + b0;
            float v1 = s1[jj] * scale + b1;
            float mx = fmaxf(v0, v1);        // row reduce across 16-lane half
#pragma unroll
            for (int msk = 1; msk < 16; msk <<= 1)
                mx = fmaxf(mx, __shfl_xor(mx, msk, 32));
            float mnew = fmaxf(mI[jj], mx);
            alpha[jj] = __expf(mI[jj] - mnew);
            float e0 = __expf(v0 - mnew);
            float e1 = __expf(v1 - mnew);
            float rs = e0 + e1;
#pragma unroll
            for (int msk = 1; msk < 16; msk <<= 1)
                rs += __shfl_xor(rs, msk, 32);
            lI[jj] = lI[jj] * alpha[jj] + rs;
            mI[jj] = mnew;
            p0[jj] = e0; p1[jj] = e1;
        }
#pragma unroll
        for (int t = 0; t < 4; t++)
#pragma unroll
            for (int jj = 0; jj < 8; jj++) oacc[t][jj] *= alpha[jj];

        // Transpose P (C-layout -> A-fragment layout) through LDS (wave-local).
#pragma unroll
        for (int jj = 0; jj < 8; jj++) {
            pw[(hi * 8 + jj) * 32 + lr]      = (bf16_t)p0[jj];
            pw[(hi * 8 + jj) * 32 + 16 + lr] = (bf16_t)p1[jj];
        }
        v16bf pf = load_a_frag(pw + (size_t)lr * 32, 0, hi);
#pragma unroll
        for (int t = 0; t < 4; t++) {
            v16bf vf = load_b_frag(Vtile + (t * 16 + lr) * VT_STRIDE, 0, hi);
            oacc[t] = wmma_bf16(pf, vf, oacc[t]);
        }
    }

    // Normalize and emit bf16 [B,N,C] (C index = h*64 + d).
#pragma unroll
    for (int t = 0; t < 4; t++) {
#pragma unroll
        for (int jj = 0; jj < 8; jj++) {
            int n = q0 + hi * 8 + jj;
            int c = h * 64 + t * 16 + lr;
            float v = oacc[t][jj] / lI[jj];
            O[((size_t)b * 2048 + n) * 1024 + c] = (bf16_t)v;
        }
    }
}

// ---------------------------------------------------------------------------
extern "C" void kernel_launch(void* const* d_in, const int* in_sizes, int n_in,
                              void* d_out, int out_size, void* d_ws, size_t ws_size,
                              hipStream_t stream) {
    const float* x      = (const float*)d_in[0];  // [2,2048,1024]
    const float* amask  = (const float*)d_in[1];  // [2,1,2048,2048]
    const float* qkv_w  = (const float*)d_in[2];  // [3072,1024]
    const float* qkv_b  = (const float*)d_in[3];  // [3072]
    const float* proj_w = (const float*)d_in[4];  // [1024,1024]
    const float* proj_b = (const float*)d_in[5];  // [1024]
    float* out = (float*)d_out;

    char* ws = (char*)d_ws;
    // Workspace layout (bytes): all bf16 unless noted. Total ~48.2 MB.
    bf16_t* xb     = (bf16_t*)(ws + 0);              //  8 MB  [4096,1024]
    bf16_t* wqkvb  = (bf16_t*)(ws + 8388608);        //  6 MB  [3072,1024]
    bf16_t* wpb    = (bf16_t*)(ws + 14680064);       //  2 MB  [1024,1024]
    float*  cosT   = (float*) (ws + 16777216);       // 64 KB  [256,64] f32
    float*  sinT   = (float*) (ws + 16842752);       // 64 KB
    bf16_t* Qb     = (bf16_t*)(ws + 16908288);       //  8 MB  [B,H,N,D]
    bf16_t* Kb     = (bf16_t*)(ws + 25296896);       //  8 MB  [B,H,N,D]
    bf16_t* Vt     = (bf16_t*)(ws + 33685504);       //  8 MB  [B,H,D,N]
    bf16_t* attnO  = (bf16_t*)(ws + 42074112);       //  8 MB  [B,N,C]

    // 0) precision staging + RoPE tables
    k_f32_to_bf16<<<4096, 256, 0, stream>>>((const float4*)x,      xb,    4194304 / 4);
    k_f32_to_bf16<<<3072, 256, 0, stream>>>((const float4*)qkv_w,  wqkvb, 3145728 / 4);
    k_f32_to_bf16<<<1024, 256, 0, stream>>>((const float4*)proj_w, wpb,   1048576 / 4);
    k_rope_tables<<<64, 256, 0, stream>>>(cosT, sinT);

    // 1) QKV GEMM + bias + RoPE + layout scatter
    k_qkv_gemm<<<dim3(24, 32), 256, 0, stream>>>(xb, wqkvb, qkv_b, cosT, sinT,
                                                 Qb, Kb, Vt);
    // 2) fused flash attention (async LDS staged K/V)
    k_flash_attn<<<dim3(16, 16, 2), 256, 0, stream>>>(Qb, Kb, Vt, amask, attnO);
    // 3) output projection -> f32
    k_proj_gemm<<<dim3(8, 32), 256, 0, stream>>>(attnO, wpb, proj_b, out);
}
